// NetConvPool_12489764897312
// MI455X (gfx1250) — matrix operand order repaired and checked by reading, not compiled, not run on hardware
//
#include <hip/hip_runtime.h>
#include <hip/hip_bf16.h>
#include <math.h>

// ---------------------------------------------------------------------------
// Sizes (fixed by the reference)
// ---------------------------------------------------------------------------
#define N0 65536
#define N1 32768
#define N2 16384
#define E0 524288
#define E1 262144
#define E2 131072
#define NIN 32
#define NLIN 64
#define HID 96
#define KMIX 4
#define ED 9

typedef __bf16  v16bf __attribute__((ext_vector_type(16)));
typedef float   v8f   __attribute__((ext_vector_type(8)));

// ---------------------------------------------------------------------------
// WMMA bf16 GEMM:  C[M,Nc] = (relu?)(A[M,Kd] @ W[Kd,Nc] + bias)
// Requirements: M % 64 == 0, Kd % 32 == 0, Nc % (16*NT) == 0.
// Block = (32,4): 4 waves; each wave computes a 16 x (16*NT) slab of C.
// Grid  = (M/64, Nc/(16*NT)).
//
// Per K-step (32 deep):
//   - block cooperatively stages the B tile [32 x 16*NT] into LDS,
//     transposed to column-major bf16 so each lane's B fragment is 16
//     contiguous bf16 (ds_load_b128-class reads, shared by all 4 waves)
//   - each wave loads its A fragment once and issues NT WMMAs against it
//
// VGPR layouts per CDNA5 ISA 7.12.2:
//   A 16x32 bf16 : lane L -> row m = L&15, half = L>>4.
//                  element e : K = half*16 + e (contiguous, not ISA table order)
//   B 32x16 bf16 : lane L -> col n = L&15 ; element e : K = half*16 + e (as A; not ISA table order)
//   C 16x16 f32  : lane L -> col n = L&15 ; vgpr r : row = half*8 + r
// ---------------------------------------------------------------------------
template <int NT>
__global__ void gemm_wmma_bf16(const float* __restrict__ A,
                               const float* __restrict__ W,
                               const float* __restrict__ bias,
                               float* __restrict__ C,
                               int M, int Kd, int Nc, int relu)
{
    constexpr int CW = 16 * NT;              // columns per block
    __shared__ __bf16 Blds[CW * 32];         // [col][k], col-major fragments

    const int lane = threadIdx.x;            // 0..31
    const int wave = threadIdx.y;            // 0..3
    const int tid  = wave * 32 + lane;       // 0..127
    const int mtile = (blockIdx.x * 4 + wave) * 16;
    const int nb    = blockIdx.y * CW;
    const int half  = lane >> 4;
    const int l15   = lane & 15;
    const int mrow  = mtile + l15;

    v8f acc[NT];
#pragma unroll
    for (int t = 0; t < NT; ++t)
        acc[t] = v8f{0.f, 0.f, 0.f, 0.f, 0.f, 0.f, 0.f, 0.f};

    for (int ko = 0; ko < Kd; ko += 32) {
        // ---- cooperative B staging: global (row-major, coalesced) ->
        //      LDS (column-major bf16) ----
#pragma unroll
        for (int i = tid; i < 32 * CW; i += 128) {
            const int k = i / CW;            // 0..31
            const int c = i % CW;            // consecutive tid -> consecutive col
            Blds[c * 32 + k] = (__bf16)W[(size_t)(ko + k) * Nc + nb + c];
        }
        __syncthreads();

        // ---- A fragment (fp32 -> bf16 on the fly), reused for NT WMMAs ----
        const float* arow = A + (size_t)mrow * Kd + ko + half * 16;
        if (ko + 32 < Kd) __builtin_prefetch(arow + 32, 0, 1);
        v16bf a;
#pragma unroll
        for (int e = 0; e < 8; ++e) a[e] = (__bf16)arow[e];
#pragma unroll
        for (int e = 0; e < 8; ++e) a[8 + e] = (__bf16)arow[8 + e];

#pragma unroll
        for (int t = 0; t < NT; ++t) {
            const __bf16* bp = &Blds[(t * 16 + l15) * 32 + half * 16];
            v16bf b;
#pragma unroll
            for (int e = 0; e < 16; ++e) b[e] = bp[e];
            acc[t] = __builtin_amdgcn_wmma_f32_16x16x32_bf16(
                /*neg_a=*/false, a, /*neg_b=*/false, b,
                /*c_mod=*/(short)0, acc[t], /*reuse_a=*/false, /*reuse_b=*/false);
        }
        __syncthreads();
    }

#pragma unroll
    for (int t = 0; t < NT; ++t) {
        const int col = nb + t * 16 + l15;
        const float bv = bias ? bias[col] : 0.0f;
#pragma unroll
        for (int r = 0; r < 8; ++r) {
            float v = acc[t][r] + bv;
            if (relu) v = fmaxf(v, 0.0f);
            C[(size_t)(mtile + half * 8 + r) * Nc + col] = v;
        }
    }
}

// ---------------------------------------------------------------------------
// Per-edge GMM message + scatter-add (mean numerator) + degree count.
// gx is [N, KMIX*HID] (k-major). Block = (32,8): one wave per edge.
// ---------------------------------------------------------------------------
__global__ void gmm_edge_kernel(const float* __restrict__ gx,
                                const int* __restrict__ src,
                                const int* __restrict__ dst,
                                const float* __restrict__ ea,
                                const float* __restrict__ mu,
                                const float* __restrict__ sigma,
                                float* __restrict__ agg,
                                float* __restrict__ cnt,
                                int E)
{
    const int e = blockIdx.x * blockDim.y + threadIdx.y;
    if (e >= E) return;
    const int lane = threadIdx.x;

    float w[KMIX];
#pragma unroll
    for (int k = 0; k < KMIX; ++k) {
        float s = 0.f;
#pragma unroll
        for (int d = 0; d < ED; ++d) {
            float diff = ea[(size_t)e * ED + d] - mu[k * ED + d];
            float sg   = sigma[k * ED + d];
            s += diff * diff / (sg * sg + 1e-15f);
        }
        w[k] = expf(-0.5f * s);
    }

    const float* g = gx + (size_t)src[e] * (KMIX * HID);
    float* a = agg + (size_t)dst[e] * HID;
#pragma unroll
    for (int t = 0; t < 3; ++t) {        // 96 outputs / 32 lanes
        const int o = lane + t * 32;
        float m = g[o]            * w[0]
                + g[HID + o]      * w[1]
                + g[2 * HID + o]  * w[2]
                + g[3 * HID + o]  * w[3];
        atomicAdd(&a[o], m);
    }
    if (lane == 0) atomicAdd(&cnt[dst[e]], 1.0f);
}

// out = relu(agg / max(cnt,1) + root)
__global__ void combine_kernel(const float* __restrict__ agg,
                               const float* __restrict__ cnt,
                               const float* __restrict__ root,
                               float* __restrict__ out, int total)
{
    const int i = blockIdx.x * blockDim.x + threadIdx.x;
    if (i >= total) return;
    const float c = fmaxf(cnt[i / HID], 1.0f);
    out[i] = fmaxf(agg[i] / c + root[i], 0.0f);
}

// scatter for pool_mean (sums + counts)
__global__ void pool_scatter_kernel(const float* __restrict__ x,
                                    const int* __restrict__ cl,
                                    float* __restrict__ s,
                                    float* __restrict__ c, int total)
{
    const int i = blockIdx.x * blockDim.x + threadIdx.x;
    if (i >= total) return;
    const int n = i / HID, o = i % HID;
    const int t = cl[n];
    atomicAdd(&s[(size_t)t * HID + o], x[i]);
    if (o == 0) atomicAdd(&c[t], 1.0f);
}

__global__ void pool_div_kernel(float* __restrict__ s,
                                const float* __restrict__ c, int total)
{
    const int i = blockIdx.x * blockDim.x + threadIdx.x;
    if (i >= total) return;
    s[i] /= fmaxf(c[i / HID], 1.0f);
}

// cat[n, 0:96] = up[map[n]] ; cat[n, 96:192] = skip[n]
__global__ void unpool_concat_kernel(const float* __restrict__ up,
                                     const int* __restrict__ map,
                                     const float* __restrict__ skip,
                                     float* __restrict__ cat, int total)
{
    const int i = blockIdx.x * blockDim.x + threadIdx.x;
    if (i >= total) return;
    const int n = i / (2 * HID), j = i % (2 * HID);
    cat[i] = (j < HID) ? up[(size_t)map[n] * HID + j]
                       : skip[(size_t)n * HID + (j - HID)];
}

// final 96 -> 2 linear (no relu)
__global__ void lin_out_kernel(const float* __restrict__ h,
                               const float* __restrict__ w,
                               const float* __restrict__ b,
                               float* __restrict__ out, int N)
{
    const int n = blockIdx.x * blockDim.x + threadIdx.x;
    if (n >= N) return;
    const float* r = h + (size_t)n * HID;
    float a0 = b[0], a1 = b[1];
#pragma unroll 8
    for (int j = 0; j < HID; ++j) {
        const float v = r[j];
        a0 += v * w[j * 2 + 0];
        a1 += v * w[j * 2 + 1];
    }
    out[n * 2 + 0] = a0;
    out[n * 2 + 1] = a1;
}

// ---------------------------------------------------------------------------
// Host-side helpers
// ---------------------------------------------------------------------------
struct GmmP { const float *Wg, *mu, *sigma, *Wroot, *b; };

static void run_gemm(const float* A, const float* W, const float* bias,
                     float* C, int M, int Kd, int Nc, int relu, hipStream_t s)
{
    if (Nc % 96 == 0) {
        gemm_wmma_bf16<6><<<dim3(M / 64, Nc / 96), dim3(32, 4), 0, s>>>(
            A, W, bias, C, M, Kd, Nc, relu);
    } else {            // Nc == 64
        gemm_wmma_bf16<4><<<dim3(M / 64, Nc / 64), dim3(32, 4), 0, s>>>(
            A, W, bias, C, M, Kd, Nc, relu);
    }
}

static void run_gmm_conv(const float* hin, int N, int Kd, const GmmP& g,
                         const int* ei, const float* ea, int E,
                         float* GX, float* ROOT, float* AGG, float* CNT,
                         float* outbuf, hipStream_t s)
{
    run_gemm(hin, g.Wg, nullptr, GX, N, Kd, KMIX * HID, 0, s);
    run_gemm(hin, g.Wroot, g.b, ROOT, N, Kd, HID, 0, s);
    hipMemsetAsync(AGG, 0, (size_t)N * HID * sizeof(float), s);
    hipMemsetAsync(CNT, 0, (size_t)N * sizeof(float), s);
    gmm_edge_kernel<<<dim3((E + 7) / 8), dim3(32, 8), 0, s>>>(
        GX, ei, ei + E, ea, g.mu, g.sigma, AGG, CNT, E);
    combine_kernel<<<(N * HID + 255) / 256, 256, 0, s>>>(AGG, CNT, ROOT, outbuf, N * HID);
}

static void run_pool(const float* xin, const int* cl, float* P, float* C,
                     int Nin, int Nout, hipStream_t s)
{
    hipMemsetAsync(P, 0, (size_t)Nout * HID * sizeof(float), s);
    hipMemsetAsync(C, 0, (size_t)Nout * sizeof(float), s);
    pool_scatter_kernel<<<(Nin * HID + 255) / 256, 256, 0, s>>>(xin, cl, P, C, Nin * HID);
    pool_div_kernel<<<(Nout * HID + 255) / 256, 256, 0, s>>>(P, C, Nout * HID);
}

extern "C" void kernel_launch(void* const* d_in, const int* in_sizes, int n_in,
                              void* d_out, int out_size, void* d_ws, size_t ws_size,
                              hipStream_t stream)
{
    // ---- graph inputs ----
    const float* x   = (const float*)d_in[0];
    const int*   ei0 = (const int*)  d_in[1];
    const float* ea0 = (const float*)d_in[2];
    const int*   c1  = (const int*)  d_in[3];
    const int*   ei1 = (const int*)  d_in[4];
    const float* ea1 = (const float*)d_in[5];
    const int*   c2  = (const int*)  d_in[6];
    const int*   ei2 = (const int*)  d_in[7];
    const float* ea2 = (const float*)d_in[8];

    // ---- params: detect flattening order by size probe ----
    auto F = [&](int i) { return (const float*)d_in[i]; };
    GmmP top0, top1, top2, down0, down1, up0, up1;
    const float *l1w, *l1b, *l2w, *l2b, *l3w, *l3b, *l4w, *l4b;
    if (in_sizes[9] == NIN * NLIN) {
        // dict insertion order; inner: Wg, mu, sigma, Wroot, b
        auto G = [&](int i) { return GmmP{F(i), F(i + 1), F(i + 2), F(i + 3), F(i + 4)}; };
        l1w = F(9);  l1b = F(10);
        top0 = G(11); top1 = G(16); top2 = G(21);
        down0 = G(26); down1 = G(31);
        up0 = G(36); up1 = G(41);
        l2w = F(46); l2b = F(47); l3w = F(48); l3b = F(49); l4w = F(50); l4b = F(51);
    } else {
        // jax sorted-key order; inner sorted: Wg, Wroot, b, mu, sigma
        auto G = [&](int i) { return GmmP{F(i), F(i + 3), F(i + 4), F(i + 1), F(i + 2)}; };
        down0 = G(9); down1 = G(14);
        l1b = F(19); l1w = F(20); l2b = F(21); l2w = F(22);
        l3b = F(23); l3w = F(24); l4b = F(25); l4w = F(26);
        top0 = G(27); top1 = G(32); top2 = G(37);
        up0 = G(42); up1 = G(47);
    }

    // ---- workspace carve-up (floats) ----
    float* w = (float*)d_ws;
    size_t off = 0;
    auto alloc = [&](size_t n) { float* p = w + off; off += n; return p; };
    float* H1   = alloc((size_t)N0 * NLIN);       // lin1 output
    float* HA   = alloc((size_t)N0 * HID);
    float* HB   = alloc((size_t)N0 * HID);
    float* GX   = alloc((size_t)N0 * KMIX * HID); // largest gx buffer (reused)
    float* ROOT = alloc((size_t)N0 * HID);
    float* AGG  = alloc((size_t)N0 * HID);
    float* CNT  = alloc((size_t)N0);
    float* P1   = alloc((size_t)N1 * HID);
    float* D1   = alloc((size_t)N1 * HID);
    float* P2   = alloc((size_t)N2 * HID);
    float* D2   = alloc((size_t)N2 * HID);
    float* C1n  = alloc((size_t)N1);
    float* C2n  = alloc((size_t)N2);
    float* CAT1 = alloc((size_t)N1 * 2 * HID);
    float* CAT0 = alloc((size_t)N0 * 2 * HID);
    (void)ws_size; (void)n_in; (void)out_size;

    // ---- forward pass ----
    // h = relu(x @ lin1 + b)                 [N0, 64]
    run_gemm(x, l1w, l1b, H1, N0, NIN, NLIN, 1, stream);

    // three GMMConv layers at level 0
    run_gmm_conv(H1, N0, NLIN, top0, ei0, ea0, E0, GX, ROOT, AGG, CNT, HA, stream);
    run_gmm_conv(HA, N0, HID,  top1, ei0, ea0, E0, GX, ROOT, AGG, CNT, HB, stream);
    run_gmm_conv(HB, N0, HID,  top2, ei0, ea0, E0, GX, ROOT, AGG, CNT, HA, stream); // skip0 = HA

    // down level 1
    run_pool(HA, c1, P1, C1n, N0, N1, stream);
    run_gmm_conv(P1, N1, HID, down0, ei1, ea1, E1, GX, ROOT, AGG, CNT, D1, stream); // skip1 = D1

    // down level 2
    run_pool(D1, c2, P2, C2n, N1, N2, stream);
    run_gmm_conv(P2, N2, HID, down1, ei2, ea2, E2, GX, ROOT, AGG, CNT, D2, stream);

    // up to level 1
    unpool_concat_kernel<<<(N1 * 2 * HID + 255) / 256, 256, 0, stream>>>(
        D2, c2, D1, CAT1, N1 * 2 * HID);
    run_gmm_conv(CAT1, N1, 2 * HID, up0, ei1, ea1, E1, GX, ROOT, AGG, CNT, P1, stream); // U1 -> P1

    // up to level 0
    unpool_concat_kernel<<<(N0 * 2 * HID + 255) / 256, 256, 0, stream>>>(
        P1, c1, HA, CAT0, N0 * 2 * HID);
    run_gmm_conv(CAT0, N0, 2 * HID, up1, ei0, ea0, E0, GX, ROOT, AGG, CNT, HB, stream);

    // head: two relu linears + final 96->2
    run_gemm(HB, l2w, l2b, HA, N0, HID, HID, 1, stream);
    run_gemm(HA, l3w, l3b, HB, N0, HID, HID, 1, stream);
    lin_out_kernel<<<(N0 + 255) / 256, 256, 0, stream>>>(HB, l4w, l4b, (float*)d_out, N0);
}
